// LearnableFeaturePermute_82042465288804
// MI455X (gfx1250) — compile-verified, matrix operand order checked by reading, not run on hardware
//
#include <hip/hip_runtime.h>
#include <hip/hip_bf16.h>
#include <cstdint>

// ---------------------------------------------------------------------------
// LearnableFeaturePermute: Sinkhorn(128x128 logits) -> P, then out = x @ P
//   x: [524288, 128] f32, out: [524288, 128] f32
// Strategy:
//   K1 (1 block): Sinkhorn in LDS, then pack P into bf16 hi/lo WMMA B-fragments
//                 (column-major per-lane layout) in workspace.
//   K2 (4096 blocks x 256 thr): split-precision bf16 GEMM via
//                 v_wmma_f32_16x16x32_bf16 (3 products: AhBh + AhBl + AlBh).
// ---------------------------------------------------------------------------

typedef __attribute__((ext_vector_type(16))) __bf16 v16bf;
typedef __attribute__((ext_vector_type(8)))  float  v8f;

#define FDIM        128
#define KTILES      4                       // 128 / 32 (K per bf16 WMMA)
#define NTILES      8                       // 128 / 16
#define FRAG_DWORDS (KTILES * NTILES * 32 * 8)   // 8192 dwords per matrix part
#define SINK_ITERS  20

// Split an f32 pair into packed bf16 hi (truncated) and bf16 lo (residual).
// K-even element goes to bits [15:0], K-odd to [31:16] (A/B VGPR pair layout).
__device__ __forceinline__ void bsplit(float a, float b, uint32_t& hi, uint32_t& lo) {
    uint32_t ba = __builtin_bit_cast(uint32_t, a);
    uint32_t bb = __builtin_bit_cast(uint32_t, b);
    uint32_t ha = ba & 0xffff0000u;
    uint32_t hb = bb & 0xffff0000u;
    hi = (ha >> 16) | hb;
    float ra = a - __builtin_bit_cast(float, ha);
    float rb = b - __builtin_bit_cast(float, hb);
    lo = (__builtin_bit_cast(uint32_t, ra) >> 16) |
         (__builtin_bit_cast(uint32_t, rb) & 0xffff0000u);
}

// ---------------------------------------------------------------------------
// Kernel 1: Sinkhorn + B-fragment packing.  One block, 128 threads (4 waves).
// ---------------------------------------------------------------------------
__global__ __launch_bounds__(128) void sinkhorn_pack_kernel(
        const float* __restrict__ logits, uint32_t* __restrict__ frag) {
    __shared__ float s[FDIM * FDIM];        // 64 KB
    const int tid = threadIdx.x;

    // Load row `tid` of logits (TEMPERATURE == 1.0 -> identity scale).
    for (int j = 0; j < FDIM; j += 4)
        *(float4*)(s + tid * FDIM + j) = *(const float4*)(logits + tid * FDIM + j);
    __syncthreads();

    for (int it = 0; it < SINK_ITERS; ++it) {
        // Row normalization: thread tid owns row tid.
        float m = -INFINITY;
        for (int j = 0; j < FDIM; ++j) m = fmaxf(m, s[tid * FDIM + j]);
        float sum = 0.f;
        for (int j = 0; j < FDIM; ++j) sum += __expf(s[tid * FDIM + j] - m);
        float lse = m + __logf(sum);
        for (int j = 0; j < FDIM; ++j) s[tid * FDIM + j] -= lse;
        __syncthreads();
        // Column normalization: thread tid owns column tid.
        m = -INFINITY;
        for (int j = 0; j < FDIM; ++j) m = fmaxf(m, s[j * FDIM + tid]);
        sum = 0.f;
        for (int j = 0; j < FDIM; ++j) sum += __expf(s[j * FDIM + tid] - m);
        lse = m + __logf(sum);
        for (int j = 0; j < FDIM; ++j) s[j * FDIM + tid] -= lse;
        __syncthreads();
    }
    for (int j = 0; j < FDIM; ++j) s[tid * FDIM + j] = __expf(s[tid * FDIM + j]);
    __syncthreads();

    // Pack P into per-lane WMMA B fragments (column-major):
    //   lane<16 : col N = nT*16+lane,   K = kb..kb+7 (v0..3), kb+16..kb+23 (v4..7), kb = kT*32
    //   lane>=16: same col, kb = kT*32+8
    // Flat dword index: ((kT*8 + nT)*32 + lane)*8 + vgpr ; lo part at +FRAG_DWORDS.
    for (int slot = tid; slot < KTILES * NTILES * 32; slot += 128) {
        const int kT   = slot >> 8;
        const int nT   = (slot >> 5) & 7;
        const int lane = slot & 31;
        const int n    = nT * 16 + (lane & 15);
        const int kb   = kT * 32 + ((lane >> 4) << 3);
        uint32_t hi[8], lo[8];
        for (int j = 0; j < 4; ++j) {
            const int k0 = kb + 2 * j;
            bsplit(s[(k0     ) * FDIM + n], s[(k0 + 1 ) * FDIM + n], hi[j],     lo[j]);
            bsplit(s[(k0 + 16) * FDIM + n], s[(k0 + 17) * FDIM + n], hi[4 + j], lo[4 + j]);
        }
        uint32_t* dh = frag + (size_t)slot * 8;
        uint32_t* dl = dh + FRAG_DWORDS;
        for (int j = 0; j < 8; ++j) { dh[j] = hi[j]; dl[j] = lo[j]; }
    }
}

// ---------------------------------------------------------------------------
// Kernel 2: split-precision bf16 WMMA GEMM.  256 threads = 8 waves; each wave
// computes a 16x128 output strip (8 accumulator tiles), 96 WMMAs per wave.
// ---------------------------------------------------------------------------
__global__ __launch_bounds__(256) void permute_gemm_kernel(
        const float* __restrict__ x, const uint32_t* __restrict__ frag,
        float* __restrict__ out, int nrows) {
    __shared__ uint32_t Bsh[2 * FRAG_DWORDS];    // 64 KB: hi frags then lo frags
    const int tid = threadIdx.x;

    // Cooperative copy of both fragment arrays into LDS (4096 x uint4).
    {
        const uint4* src = (const uint4*)frag;
        uint4*       dst = (uint4*)Bsh;
        for (int i = tid; i < (2 * FRAG_DWORDS) / 4; i += 256) dst[i] = src[i];
    }
    __syncthreads();

    const int wave    = tid >> 5;
    const int lane    = tid & 31;
    const int rowBase = (blockIdx.x * 8 + wave) * 16;
    if (rowBase + 16 > nrows) return;

    const int    halfSel = lane >> 4;                       // 0 or 1
    const int    rowA    = rowBase + (lane & 15);
    const float* xr      = x + (size_t)rowA * FDIM + halfSel * 8;

    v8f acc[NTILES];
    for (int n = 0; n < NTILES; ++n) acc[n] = v8f{0.f, 0.f, 0.f, 0.f, 0.f, 0.f, 0.f, 0.f};

    for (int kT = 0; kT < KTILES; ++kT) {
        const float* p = xr + kT * 32;
        if (kT + 1 < KTILES) __builtin_prefetch(p + 32, 0, 1);  // global_prefetch_b8
        const float4 f0 = *(const float4*)(p);
        const float4 f1 = *(const float4*)(p + 4);
        const float4 g0 = *(const float4*)(p + 16);
        const float4 g1 = *(const float4*)(p + 20);

        union { uint32_t u[8]; v16bf v; } Ah, Al;
        bsplit(f0.x, f0.y, Ah.u[0], Al.u[0]);
        bsplit(f0.z, f0.w, Ah.u[1], Al.u[1]);
        bsplit(f1.x, f1.y, Ah.u[2], Al.u[2]);
        bsplit(f1.z, f1.w, Ah.u[3], Al.u[3]);
        bsplit(g0.x, g0.y, Ah.u[4], Al.u[4]);
        bsplit(g0.z, g0.w, Ah.u[5], Al.u[5]);
        bsplit(g1.x, g1.y, Ah.u[6], Al.u[6]);
        bsplit(g1.z, g1.w, Ah.u[7], Al.u[7]);

        for (int nT = 0; nT < NTILES; ++nT) {
            union { uint32_t u[8]; v16bf v; } Bh, Bl;
            const uint32_t* bh = &Bsh[(((kT * NTILES + nT) * 32) + lane) * 8];
            const uint32_t* bl = bh + FRAG_DWORDS;
            *(uint4*)&Bh.u[0] = *(const uint4*)(bh);
            *(uint4*)&Bh.u[4] = *(const uint4*)(bh + 4);
            *(uint4*)&Bl.u[0] = *(const uint4*)(bl);
            *(uint4*)&Bl.u[4] = *(const uint4*)(bl + 4);

            // out += Ah*Bh + Ah*Bl + Al*Bh   (fp32-accurate bf16x3 emulation)
            acc[nT] = __builtin_amdgcn_wmma_f32_16x16x32_bf16(
                false, Ah.v, false, Bh.v, (short)0, acc[nT], false, false);
            acc[nT] = __builtin_amdgcn_wmma_f32_16x16x32_bf16(
                false, Ah.v, false, Bl.v, (short)0, acc[nT], false, false);
            acc[nT] = __builtin_amdgcn_wmma_f32_16x16x32_bf16(
                false, Al.v, false, Bh.v, (short)0, acc[nT], false, false);
        }
    }

    // C/D layout: VGPR i -> row (rowBase + i + 8*halfSel), col = nT*16 + (lane&15).
    const int rOut  = rowBase + (halfSel << 3);
    const int cBase = lane & 15;
    for (int nT = 0; nT < NTILES; ++nT)
        for (int i = 0; i < 8; ++i)
            out[(size_t)(rOut + i) * FDIM + nT * 16 + cBase] = acc[nT][i];
}

// ---------------------------------------------------------------------------
extern "C" void kernel_launch(void* const* d_in, const int* in_sizes, int n_in,
                              void* d_out, int out_size, void* d_ws, size_t ws_size,
                              hipStream_t stream) {
    const float* x      = (const float*)d_in[0];
    const float* logits = (const float*)d_in[1];
    float*       out    = (float*)d_out;
    uint32_t*    frag   = (uint32_t*)d_ws;     // needs 2*FRAG_DWORDS*4 = 64 KB

    const int nrows = in_sizes[0] / FDIM;      // 524288

    sinkhorn_pack_kernel<<<1, 128, 0, stream>>>(logits, frag);

    const int blocks = (nrows + 127) / 128;    // 8 waves x 16 rows per block
    permute_gemm_kernel<<<blocks, 256, 0, stream>>>(x, frag, out, nrows);
}
